// SelfAttention_22119081574645
// MI455X (gfx1250) — matrix operand order repaired and checked by reading, not compiled, not run on hardware
//
#include <hip/hip_runtime.h>

// Fixed problem shape
#define BB   4
#define SS   2048
#define HH   768
#define NHH  12
#define DHH  64
#define KT   9
#define MM   (BB*SS)      // 8192
#define NK   (NHH*KT)     // 108
#define NKP  128          // padded kern width
#define PAD  ((KT-1)/2)   // 4

typedef __attribute__((ext_vector_type(16))) __bf16 bf16x16;
typedef __attribute__((ext_vector_type(8)))  float  f32x8;

#define BDIM 256
#define BM   64
#define BN   128
#define BK   32
#define BKPA 40    // padded LDS row stride (bf16): 80B rows -> 16B-aligned, 20-bank step
#define BKPB 40

__device__ __forceinline__ __bf16 to_bf16(float f) { return (__bf16)f; }

union Frag  { bf16x16 v; int4 q[2]; };
union Pack8 { __bf16 h[8];  int4 q;    };
union Pack16{ __bf16 h[16]; int4 q[2]; };

// D[M,N] = A[M,768] * W[768,N] + bias[N];  MODE 1: D *= extra (same layout as D)
template<int N, int MODE>
__global__ __launch_bounds__(BDIM)
void gemm_wmma(const float* __restrict__ A, const float* __restrict__ W,
               const float* __restrict__ bias, const float* __restrict__ extra,
               float* __restrict__ out)
{
    constexpr int K = HH;
    __shared__ __align__(16) __bf16 sA[2][BM * BKPA];
    __shared__ __align__(16) __bf16 sB[2][BN * BKPB];

    const int tid   = threadIdx.x;
    const int lane  = tid & 31;
    const int wave  = tid >> 5;
    const int m0    = (wave & 3) * 16;     // 4 waves along M
    const int n0    = (wave >> 2) * 64;    // 2 waves along N
    const int lm    = lane & 15;
    const int hi    = lane >> 4;

    const int blockN0 = blockIdx.x * BN;
    const int blockM0 = blockIdx.y * BM;

    // staging addresses (constant per thread)
    const int aRow = tid >> 2;             // 0..63
    const int aKc  = (tid & 3) * 8;        // 0,8,16,24
    const int bN   = tid & 127;            // column within B tile
    const int bK0  = (tid >> 7) * 16;      // 0 or 16

    // fragment base element offsets (constant per thread)
    const int aBase = (m0 + lm) * BKPA + (hi ? 8 : 0);   // + {0,16} halves -> 16B aligned
    const int bBase = (hi ? 8 : 0) + lm * BKPB;          // + f*16*BKPB     -> 16B aligned

    f32x8 acc[4] = {};

    auto stage = [&](int buf, int kt) {
        // ---- A tile: 8 contiguous k per thread, one b128 store
        {
            const float4* src = reinterpret_cast<const float4*>(
                A + (size_t)(blockM0 + aRow) * K + kt + aKc);
            float4 f0 = src[0], f1 = src[1];
            Pack8 p;
            p.h[0]=to_bf16(f0.x); p.h[1]=to_bf16(f0.y); p.h[2]=to_bf16(f0.z); p.h[3]=to_bf16(f0.w);
            p.h[4]=to_bf16(f1.x); p.h[5]=to_bf16(f1.y); p.h[6]=to_bf16(f1.z); p.h[7]=to_bf16(f1.w);
            *reinterpret_cast<int4*>(&sA[buf][aRow * BKPA + aKc]) = p.q;
        }
        // ---- B tile transposed: one column (16 k) per thread, two b128 stores
        {
            const float* src = W + (size_t)(kt + bK0) * N + blockN0 + bN;
            Pack16 p;
            #pragma unroll
            for (int k = 0; k < 16; ++k) p.h[k] = to_bf16(src[(size_t)k * N]);
            int4* dst = reinterpret_cast<int4*>(&sB[buf][bN * BKPB + bK0]);
            dst[0] = p.q[0];
            dst[1] = p.q[1];
        }
    };

    auto compute = [&](int buf) {
        // A fragment 16x32 (ISA 7.12.2): lanes0-15 K=0..7,16..23 ; lanes16-31 K=8..15,24..31
        Frag a;
        a.q[0] = *reinterpret_cast<const int4*>(&sA[buf][aBase]);
        a.q[1] = *reinterpret_cast<const int4*>(&sA[buf][aBase + 16]);
        // 4 B fragments 32x16: lanes0-15 K=0..7,16..23 ; lanes16-31 K=8..15,24..31
        Frag b[4];
        #pragma unroll
        for (int f = 0; f < 4; ++f) {
            const __bf16* p = &sB[buf][bBase + (n0 + f*16) * BKPB];
            b[f].q[0] = *reinterpret_cast<const int4*>(p);
            b[f].q[1] = *reinterpret_cast<const int4*>(p + 16);
        }
        #pragma unroll
        for (int f = 0; f < 4; ++f)
            acc[f] = __builtin_amdgcn_wmma_f32_16x16x32_bf16(
                false, a.v, false, b[f].v, (short)0, acc[f], false, false);
    };

    stage(0, 0);
    int it = 0;
    for (int kt = 0; kt < K; kt += BK, ++it) {
        const int cur = it & 1;
        __syncthreads();
        if (kt + BK < K) stage(cur ^ 1, kt + BK);   // overlap next tile with WMMA
        compute(cur);
    }

    // epilogue: acc[f][r] -> row m0+r+8*hi, col n0+f*16+(lane&15)
    #pragma unroll
    for (int f = 0; f < 4; ++f) {
        const int col = blockN0 + n0 + f*16 + lm;
        const float bv = bias[col];
        #pragma unroll
        for (int r = 0; r < 8; ++r) {
            const int row = blockM0 + m0 + r + 8*hi;
            const size_t idx = (size_t)row * N + col;
            float v = acc[f][r] + bv;
            if (MODE == 1) v *= extra[idx];
            out[idx] = v;
        }
    }
}

// Zero-pad Wk [768,108] -> wsWk [768,128], bk [108] -> wsBk [128]
__global__ void pad_wk(const float* __restrict__ Wk, const float* __restrict__ bk,
                       float* __restrict__ wsWk, float* __restrict__ wsBk)
{
    int idx = blockIdx.x * blockDim.x + threadIdx.x;
    if (idx < HH * NKP) {
        int r = idx >> 7, c = idx & (NKP - 1);
        wsWk[idx] = (c < NK) ? Wk[r * NK + c] : 0.0f;
    }
    if (idx < NKP) wsBk[idx] = (idx < NK) ? bk[idx] : 0.0f;
}

// depth[b,s,h] = sum_i x[b,s+i-4,h] * dw[i,0,h]   (float4 over h)
__global__ void depthwise_conv(const float* __restrict__ x, const float* __restrict__ dw,
                               float* __restrict__ depth)
{
    int idx = blockIdx.x * blockDim.x + threadIdx.x;   // MM * (HH/4)
    if (idx >= MM * (HH/4)) return;
    int h = (idx % (HH/4)) * 4;
    int m = idx / (HH/4);
    int s = m % SS, b = m / SS;
    float ax = 0.f, ay = 0.f, az = 0.f, aw = 0.f;
    #pragma unroll
    for (int i = 0; i < KT; ++i) {
        int sp = s + i - PAD;
        if (sp >= 0 && sp < SS) {
            float4 xv = *reinterpret_cast<const float4*>(x + ((size_t)(b*SS + sp))*HH + h);
            float4 wv = *reinterpret_cast<const float4*>(dw + (size_t)i*HH + h);
            ax += xv.x * wv.x; ay += xv.y * wv.y; az += xv.z * wv.z; aw += xv.w * wv.w;
        }
    }
    *reinterpret_cast<float4*>(depth + (size_t)m*HH + h) = make_float4(ax, ay, az, aw);
}

// softmax over 9 taps per (row, head), in place on [M, 128] padded
__global__ void tap_softmax(float* __restrict__ kern)
{
    int idx = blockIdx.x * blockDim.x + threadIdx.x;   // over M*NH
    if (idx >= MM * NHH) return;
    float* p = kern + (size_t)(idx / NHH) * NKP + (idx % NHH) * KT;
    float mx = p[0];
    #pragma unroll
    for (int i = 1; i < KT; ++i) mx = fmaxf(mx, p[i]);
    float e[KT], s = 0.f;
    #pragma unroll
    for (int i = 0; i < KT; ++i) { e[i] = __expf(p[i] - mx); s += e[i]; }
    float inv = 1.f / s;
    #pragma unroll
    for (int i = 0; i < KT; ++i) p[i] = e[i] * inv;
}

// out[b,s,h] = sum_k cout[b,s+k-4,h] * kern[m, (h/64)*9 + k]   (float4 over h)
__global__ void dynconv_out(const float* __restrict__ cout, const float* __restrict__ kern,
                            float* __restrict__ out)
{
    int idx = blockIdx.x * blockDim.x + threadIdx.x;   // MM * (HH/4)
    if (idx >= MM * (HH/4)) return;
    int h = (idx % (HH/4)) * 4;
    int m = idx / (HH/4);
    int s = m % SS, b = m / SS;
    const float* kp = kern + (size_t)m * NKP + (h >> 6) * KT;
    float ax = 0.f, ay = 0.f, az = 0.f, aw = 0.f;
    #pragma unroll
    for (int k = 0; k < KT; ++k) {
        int sp = s + k - PAD;
        if (sp >= 0 && sp < SS) {
            float4 cv = *reinterpret_cast<const float4*>(cout + ((size_t)(b*SS + sp))*HH + h);
            float w = kp[k];
            ax += cv.x * w; ay += cv.y * w; az += cv.z * w; aw += cv.w * w;
        }
    }
    *reinterpret_cast<float4*>(out + (size_t)m*HH + h) = make_float4(ax, ay, az, aw);
}

extern "C" void kernel_launch(void* const* d_in, const int* in_sizes, int n_in,
                              void* d_out, int out_size, void* d_ws, size_t ws_size,
                              hipStream_t stream) {
    const float* x     = (const float*)d_in[0];
    // d_in[1] = attn_mask (all ones, unused by reference math)
    const float* Wq    = (const float*)d_in[2];
    const float* bq    = (const float*)d_in[3];
    const float* dw    = (const float*)d_in[4];
    const float* pw    = (const float*)d_in[5];
    const float* bconv = (const float*)d_in[6];
    const float* Wk    = (const float*)d_in[7];
    const float* bk    = (const float*)d_in[8];
    const float* Wp    = (const float*)d_in[9];
    const float* bp    = (const float*)d_in[10];
    float* out = (float*)d_out;

    const size_t MH = (size_t)MM * HH;               // 6,291,456
    float* ws0  = (float*)d_ws;                      // q, then conv_attn   [M,768]
    float* ws1  = ws0 + MH;                          // depth, then cout    [M,768]
    float* ws2  = ws1 + MH;                          // kern (padded)       [M,128]
    float* wsWk = ws2 + (size_t)MM * NKP;            // padded Wk           [768,128]
    float* wsBk = wsWk + (size_t)HH * NKP;           // padded bk           [128]

    dim3 blk(BDIM);
    dim3 grdH(HH / BN, MM / BM);                     // (6, 128)
    dim3 grdK(1,       MM / BM);                     // (1, 128)
    const int ewBlocks = (MM * (HH/4) + BDIM - 1) / BDIM;

    // 0) pad Wk/bk into workspace
    pad_wk<<<(HH * NKP + BDIM - 1) / BDIM, blk, 0, stream>>>(Wk, bk, wsWk, wsBk);
    // 1) q = x*Wq + bq
    gemm_wmma<HH, 0><<<grdH, blk, 0, stream>>>(x, Wq, bq, nullptr, ws0);
    // 2) depthwise conv
    depthwise_conv<<<ewBlocks, blk, 0, stream>>>(x, dw, ws1);
    // 3) conv_attn = (depth*pw + bconv) * q
    gemm_wmma<HH, 1><<<grdH, blk, 0, stream>>>(ws1, pw, bconv, ws0, ws0);
    // 4) cout = x*Wp + bp
    gemm_wmma<HH, 0><<<grdH, blk, 0, stream>>>(x, Wp, bp, nullptr, ws1);
    // 5) logits = conv_attn*WkPadded + bkPadded   [M,128]
    gemm_wmma<NKP, 0><<<grdK, blk, 0, stream>>>(ws0, wsWk, wsBk, nullptr, ws2);
    // 6) softmax over 9 taps
    tap_softmax<<<(MM * NHH + BDIM - 1) / BDIM, blk, 0, stream>>>(ws2);
    // 7) tap-weighted unfold
    dynconv_out<<<ewBlocks, blk, 0, stream>>>(ws1, ws2, out);
}
